// SubcenterArcFace_82085414961614
// MI455X (gfx1250) — matrix operand order repaired and checked by reading, not compiled, not run on hardware
//
#include <hip/hip_runtime.h>
#include <hip/hip_bf16.h>
#include <math.h>

// ---- problem constants ----
#define EMBED_DIM    512
#define NUM_CLASSES  100000
#define NSUB         3
#define BATCH        256
#define SCALE_F      64.0f
// margin = 0.5
#define COS_M   0.87758255f      // cos(0.5)
#define SIN_M   0.47942555f      // sin(0.5)
#define TH_F   (-0.87758255f)    // cos(pi-0.5)
#define MM_F    0.23971277f      // sin(pi-0.5)*0.5

// ---- tiling ----
#define CLS_PER_BLK 16
#define ROWS        (CLS_PER_BLK * NSUB)   // 48 weight rows per block
#define LDW         (EMBED_DIM + 16)       // bf16 row stride in LDS: 528 -> 1056B, 32B aligned
#define NWAVES      6
#define BLK_THREADS (NWAVES * 32)          // 192

typedef __attribute__((ext_vector_type(16))) __bf16 v16bf;
typedef __attribute__((ext_vector_type(8)))  __bf16 v8bf;
typedef __attribute__((ext_vector_type(4)))  __bf16 v4bf;
typedef __attribute__((ext_vector_type(8)))  float  v8f;
typedef __attribute__((ext_vector_type(4)))  float  v4f;

// -------- kernel 1: normalize embeddings -> bf16, zero accumulators --------
__global__ __launch_bounds__(256) void norm_emb_kernel(
    const float* __restrict__ emb, __bf16* __restrict__ eB,
    float* __restrict__ gsum, float* __restrict__ gll)
{
    const int row = blockIdx.x;       // 0..255
    const int tid = threadIdx.x;      // 0..255

    if (blockIdx.x == 0) {            // zero global accumulators (harness poisons ws)
        gsum[tid] = 0.0f;
        gll[tid]  = 0.0f;
    }

    const float v0 = emb[row * EMBED_DIM + tid];
    const float v1 = emb[row * EMBED_DIM + tid + 256];
    float ss = v0 * v0 + v1 * v1;
    #pragma unroll
    for (int m = 16; m >= 1; m >>= 1) ss += __shfl_xor(ss, m, 32);

    __shared__ float tot;
    if (tid == 0) tot = 0.0f;
    __syncthreads();
    if ((tid & 31) == 0) atomicAdd(&tot, ss);
    __syncthreads();

    const float sc = 1.0f / fmaxf(sqrtf(tot), 1e-12f);
    eB[row * EMBED_DIM + tid]       = (__bf16)(v0 * sc);
    eB[row * EMBED_DIM + tid + 256] = (__bf16)(v1 * sc);
}

// -------- kernel 2: fused normalize-W + bf16 WMMA GEMM + subcenter/margin/expsum ----
__global__ __launch_bounds__(BLK_THREADS) void arc_main_kernel(
    const __bf16* __restrict__ eB,      // 256 x 512 bf16, row major (normalized)
    const float*  __restrict__ W,       // 300000 x 512 f32
    const int*    __restrict__ labels,  // 256
    float* __restrict__ gsum,           // 256 running sum(exp(logit-64))
    float* __restrict__ gll)            // 256 label logits
{
    __shared__ __bf16 sB[ROWS * LDW];   // 50688 B; later reused as float cos[256][48]
    __shared__ float  sSum[BATCH];
    __shared__ int    sLab[BATCH];

    const int tid  = threadIdx.x;
    const int lane = tid & 31;
    const int wid  = tid >> 5;              // 0..5
    const long rowBase = (long)blockIdx.x * ROWS;

    for (int i = tid; i < BATCH; i += BLK_THREADS) {
        sSum[i] = 0.0f;
        sLab[i] = labels[i];
    }

    // ---- stage 1: load 8 weight rows per wave (b128), normalize, store bf16 (b64) ----
    #pragma unroll 2
    for (int rr = 0; rr < ROWS / NWAVES; ++rr) {
        const int r = wid * (ROWS / NWAVES) + rr;
        const float* wp = W + (rowBase + r) * (long)EMBED_DIM;
        v4f v[4];
        #pragma unroll
        for (int j = 0; j < 4; ++j)         // 512B contiguous per instruction
            v[j] = *(const v4f*)(wp + j * 128 + lane * 4);
        float ss = 0.0f;
        #pragma unroll
        for (int j = 0; j < 4; ++j)
            #pragma unroll
            for (int t = 0; t < 4; ++t) ss += v[j][t] * v[j][t];
        #pragma unroll
        for (int m = 16; m >= 1; m >>= 1) ss += __shfl_xor(ss, m, 32);
        const float sc = 1.0f / fmaxf(sqrtf(ss), 1e-12f);
        #pragma unroll
        for (int j = 0; j < 4; ++j) {
            v4bf b;
            #pragma unroll
            for (int t = 0; t < 4; ++t) b[t] = (__bf16)(v[j][t] * sc);
            *(v4bf*)&sB[r * LDW + j * 128 + lane * 4] = b;   // 8B store, aligned
        }
    }
    __syncthreads();

    // ---- stage 2: WMMA GEMM: D[m][n] = sum_k e[m][k] * w[n][k] ----
    const int ntile = wid % 3;          // which 16-row group of the 48
    const int mhalf = wid / 3;          // batch halves 0..127 / 128..255
    const int mlane = lane & 15;
    const int hi16  = lane >> 4;        // 0 for lanes 0-15, 1 for lanes 16-31

    v8f acc[8] = {};                    // 8 M-tiles of 16x16 f32

    #pragma unroll 1
    for (int kc = 0; kc < EMBED_DIM / 32; ++kc) {
        const int kb = kc * 32;
        // B fragment: lane<16 -> K kb..kb+15 of row ntile*16+mlane; lane>=16 -> K kb+16..kb+31
        const __bf16* bp = &sB[(ntile * 16 + mlane) * LDW + kb + hi16 * 16];
        const v16bf bf = *(const v16bf*)bp;     // 32B, aligned (LDW*2 = 1056 = 33*32)

        #pragma unroll
        for (int mt = 0; mt < 8; ++mt) {
            const int m = mhalf * 128 + mt * 16 + mlane;
            // A fragment (contiguous, matches B): lane<16 holds K kb..kb+15, lane>=16 K kb+16..kb+31
            const __bf16* ap = eB + m * EMBED_DIM + kb + hi16 * 16;
            const v8bf lo = *(const v8bf*)ap;
            const v8bf hh = *(const v8bf*)(ap + 8);
            const v16bf af = __builtin_shufflevector(lo, hh,
                0, 1, 2, 3, 4, 5, 6, 7, 8, 9, 10, 11, 12, 13, 14, 15);
            acc[mt] = __builtin_amdgcn_wmma_f32_16x16x32_bf16(
                false, af, false, bf, (short)0, acc[mt], false, false);
        }
    }
    __syncthreads();   // all waves done reading sB before it is reused

    // ---- stage 3: spill cosines to LDS (reuse sB as float[256][48]) ----
    float* sCos = (float*)sB;           // 12672 floats available >= 256*48
    {
        const int n    = ntile * 16 + mlane;
        const int mofs = 8 * hi16;      // D layout: VGPR r -> M = r + 8*(lane>>4)
        #pragma unroll
        for (int mt = 0; mt < 8; ++mt) {
            const int mb = mhalf * 128 + mt * 16 + mofs;
            #pragma unroll
            for (int r = 0; r < 8; ++r)
                sCos[(mb + r) * ROWS + n] = acc[mt][r];
        }
    }
    __syncthreads();

    // ---- stage 4: subcenter max, margin at label, partial sum(exp(logit-64)) ----
    #pragma unroll 1
    for (int idx = tid; idx < BATCH * CLS_PER_BLK; idx += BLK_THREADS) {
        const int m = idx >> 4;
        const int c = idx & 15;
        const float* p = &sCos[m * ROWS + 3 * c];
        const float cm = fmaxf(p[0], fmaxf(p[1], p[2]));
        const int C = blockIdx.x * CLS_PER_BLK + c;
        float logit;
        if (C == sLab[m]) {
            const float sine = sqrtf(fmaxf(0.0f, 1.0f - cm * cm));
            const float phi  = cm * COS_M - sine * SIN_M;
            const float sel  = (cm > TH_F) ? phi : (cm - MM_F);
            logit = SCALE_F * sel;
            gll[m] = logit;             // exactly one block writes per m
        } else {
            logit = SCALE_F * cm;
        }
        atomicAdd(&sSum[m], __expf(logit - SCALE_F));  // max logit <= SCALE
    }
    __syncthreads();

    for (int m = tid; m < BATCH; m += BLK_THREADS)
        atomicAdd(&gsum[m], sSum[m]);
}

// -------- kernel 3: mean cross entropy --------
__global__ __launch_bounds__(256) void finalize_kernel(
    const float* __restrict__ gsum, const float* __restrict__ gll,
    float* __restrict__ out)
{
    const int tid = threadIdx.x;
    float li = logf(gsum[tid]) + SCALE_F - gll[tid];   // logsumexp - label_logit
    #pragma unroll
    for (int m = 16; m >= 1; m >>= 1) li += __shfl_xor(li, m, 32);
    __shared__ float part[8];
    if ((tid & 31) == 0) part[tid >> 5] = li;
    __syncthreads();
    if (tid == 0) {
        float t = 0.0f;
        #pragma unroll
        for (int i = 0; i < 8; ++i) t += part[i];
        out[0] = t / (float)BATCH;
    }
}

extern "C" void kernel_launch(void* const* d_in, const int* in_sizes, int n_in,
                              void* d_out, int out_size, void* d_ws, size_t ws_size,
                              hipStream_t stream) {
    const float* emb    = (const float*)d_in[0];   // (256, 512) f32
    const int*   labels = (const int*)d_in[1];     // (256,) int
    const float* W      = (const float*)d_in[2];   // (300000, 512) f32
    float* out = (float*)d_out;

    // workspace layout: [0..255] gsum f32, [256..511] gll f32, then bf16 embeddings
    float*  gsum = (float*)d_ws;
    float*  gll  = gsum + 256;
    __bf16* eB   = (__bf16*)((char*)d_ws + 2048);  // 256*512*2 = 256 KB

    norm_emb_kernel<<<BATCH, 256, 0, stream>>>(emb, eB, gsum, gll);
    arc_main_kernel<<<NUM_CLASSES / CLS_PER_BLK, BLK_THREADS, 0, stream>>>(
        eB, W, labels, gsum, gll);
    finalize_kernel<<<1, 256, 0, stream>>>(gsum, gll, out);
    (void)in_sizes; (void)n_in; (void)out_size; (void)ws_size;
}
